// GenScore_GGCN_34144990003644
// MI455X (gfx1250) — matrix operand order repaired and checked by reading, not compiled, not run on hardware
//
#include <hip/hip_runtime.h>

#define B_   16
#define NL_  48
#define NT_  400
#define APR_ 24
#define H_   128
#define NLT_ (B_*NL_)        // 768 ligand nodes
#define NTT_ (B_*NT_)        // 6400 protein nodes
#define LE_  (B_*NL_*4)      // 3072 ligand edges
#define PE_  (B_*NT_*12)     // 76800 protein edges
#define LOG2PI_ 1.8378770664093453f

typedef __attribute__((ext_vector_type(16))) _Float16 v16h;
typedef __attribute__((ext_vector_type(8)))  float    v8f;
typedef __attribute__((ext_vector_type(4)))  float    f4;

// ---------------- utility ----------------
__global__ __launch_bounds__(256) void k_fill0(float* p, int n) {
  int i = blockIdx.x * 256 + threadIdx.x;
  if (i < n) p[i] = 0.f;
}

// transpose+convert nmat 128x128 f32 matrices: Wt[m][n][k] = (f16)W[m][k][n]
__global__ __launch_bounds__(256) void k_prep_wt(const float* __restrict__ W,
                                                 _Float16* __restrict__ Wt, int total) {
  int i = blockIdx.x * 256 + threadIdx.x;
  if (i >= total) return;
  int m = i >> 14, r = i & 16383, n = r >> 7, k = r & 127;
  Wt[i] = (_Float16)W[(m << 14) + (k << 7) + n];
}

// pack pi|sigma|mu weights into padded transposed [32][128] f16
__global__ __launch_bounds__(256) void k_prep_head(const float* __restrict__ pw,
                                                   const float* __restrict__ sw,
                                                   const float* __restrict__ mw,
                                                   _Float16* __restrict__ Wt) {
  int i = blockIdx.x * 256 + threadIdx.x;
  if (i >= 32 * 128) return;
  int n = i >> 7, k = i & 127;
  float v = 0.f;
  if (n < 10)       v = pw[k * 10 + n];
  else if (n < 20)  v = sw[k * 10 + (n - 10)];
  else if (n < 30)  v = mw[k * 10 + (n - 20)];
  Wt[i] = (_Float16)v;
}

// small-K encoder GEMM: Y[M,128] = X[M,K] @ W[K,128] + b
__global__ __launch_bounds__(256) void k_enc(const float* __restrict__ X,
                                             const float* __restrict__ W,
                                             const float* __restrict__ bias,
                                             float* __restrict__ Y, int M, int K) {
  int i = blockIdx.x * 256 + threadIdx.x;
  if (i >= M * 128) return;
  int m = i >> 7, f = i & 127;
  float s = bias[f];
  for (int k = 0; k < K; ++k) s += X[m * K + k] * W[k * 128 + f];
  Y[i] = s;
}

// ---------------- WMMA 128-K GEMM: Y[M,128] = X[M,128] @ Wt' + bias ----------------
// Wt is f16, transposed [n][k]. 8 waves/block, each wave owns a 16-wide N tile.
__global__ __launch_bounds__(256) void k_gemm128(const float* __restrict__ X,
                                                 const _Float16* __restrict__ Wt,
                                                 const float* __restrict__ bias,
                                                 float* __restrict__ Y) {
  int lane = threadIdx.x & 31, wv = threadIdx.x >> 5;
  int hf = lane >> 4, l16 = lane & 15;
  int mtile = blockIdx.x;
  const float* xr = X + (size_t)(mtile * 16 + l16) * 128;
  const _Float16* wr = Wt + (size_t)(wv * 16 + l16) * 128;
  v8f acc = {};
#pragma unroll
  for (int k0 = 0; k0 < 128; k0 += 32) {
    // A fragment (16-bit A 16x32): elem e -> K = k0 + 16h + e (matches B)
    f4 x0 = *(const f4*)(xr + k0 + 16 * hf);
    f4 x1 = *(const f4*)(xr + k0 + 16 * hf + 4);
    f4 x2 = *(const f4*)(xr + k0 + 16 * hf + 8);
    f4 x3 = *(const f4*)(xr + k0 + 16 * hf + 12);
    v16h a;
    a[0] = (_Float16)x0[0]; a[1] = (_Float16)x0[1]; a[2] = (_Float16)x0[2]; a[3] = (_Float16)x0[3];
    a[4] = (_Float16)x1[0]; a[5] = (_Float16)x1[1]; a[6] = (_Float16)x1[2]; a[7] = (_Float16)x1[3];
    a[8] = (_Float16)x2[0]; a[9] = (_Float16)x2[1]; a[10] = (_Float16)x2[2]; a[11] = (_Float16)x2[3];
    a[12] = (_Float16)x3[0]; a[13] = (_Float16)x3[1]; a[14] = (_Float16)x3[2]; a[15] = (_Float16)x3[3];
    // B fragment (16-bit B 32x16): elem e -> K = k0 + 16h + e, contiguous in Wt
    v16h bfr = *(const v16h*)(wr + k0 + 16 * hf);
    acc = __builtin_amdgcn_wmma_f32_16x16x32_f16(false, a, false, bfr, (short)0, acc, false, false);
  }
  int col = wv * 16 + l16;
  float bi = bias[col];
#pragma unroll
  for (int r = 0; r < 8; ++r)
    Y[(size_t)(mtile * 16 + 8 * hf + r) * 128 + col] = acc[r] + bi;
}

// ---------------- GGCN per-edge gate + segment sums ----------------
__global__ __launch_bounds__(256) void k_edge(const float* __restrict__ Bx,
                                              const float* __restrict__ Dx,
                                              const float* __restrict__ Ex,
                                              float* __restrict__ Ce,
                                              const int* __restrict__ ei, int E,
                                              float* __restrict__ num,
                                              float* __restrict__ den) {
  int i = blockIdx.x * 256 + threadIdx.x;
  if (i >= E * 128) return;
  int e = i >> 7, f = i & 127;
  int s = ei[e], d = ei[E + e];
  float v = Dx[d * 128 + f] + Ex[s * 128 + f] + Ce[i];
  float sg = 1.f / (1.f + expf(-v));
  atomicAdd(&num[d * 128 + f], sg * Bx[s * 128 + f]);
  atomicAdd(&den[d * 128 + f], sg);
  Ce[i] = v;  // e_ij (in place)
}

__global__ __launch_bounds__(256) void k_combine(float* __restrict__ A,
                                                 const float* __restrict__ num,
                                                 const float* __restrict__ den, int n) {
  int i = blockIdx.x * 256 + threadIdx.x;
  if (i < n) A[i] += num[i] / (den[i] + 1e-6f);
}

// per-feature batch stats over R rows
__global__ __launch_bounds__(256) void k_bnstats(const float* __restrict__ Z, int R,
                                                 float* __restrict__ mean,
                                                 float* __restrict__ rstd) {
  __shared__ float s1[256], s2[256];
  int f = blockIdx.x;
  float a = 0.f, b = 0.f;
  for (int r = threadIdx.x; r < R; r += 256) {
    float v = Z[(size_t)r * 128 + f]; a += v; b += v * v;
  }
  s1[threadIdx.x] = a; s2[threadIdx.x] = b;
  __syncthreads();
  for (int o = 128; o > 0; o >>= 1) {
    if (threadIdx.x < o) { s1[threadIdx.x] += s1[threadIdx.x + o]; s2[threadIdx.x] += s2[threadIdx.x + o]; }
    __syncthreads();
  }
  if (threadIdx.x == 0) {
    float m = s1[0] / (float)R;
    float v = s2[0] / (float)R - m * m;
    mean[f] = m; rstd[f] = rsqrtf(v + 1e-5f);
  }
}

__global__ __launch_bounds__(256) void k_bnres(float* __restrict__ X, const float* __restrict__ Q,
                                               const float* __restrict__ mean,
                                               const float* __restrict__ rstd,
                                               const float* __restrict__ g,
                                               const float* __restrict__ bb, int n) {
  int i = blockIdx.x * 256 + threadIdx.x;
  if (i >= n) return;
  int f = i & 127;
  float t = (Q[i] - mean[f]) * rstd[f] * g[f] + bb[f];
  X[i] += (t > 0.f) ? t : 0.f;
}

// ---------------- pair BN statistics (analytic over cross product) ----------------
__global__ __launch_bounds__(256) void k_colsum(const float* __restrict__ Z,
                                                float* __restrict__ s, float* __restrict__ s2,
                                                int rowsPerB) {
  int i = blockIdx.x * 256 + threadIdx.x;
  if (i >= B_ * 128) return;
  int b = i >> 7, f = i & 127;
  const float* p = Z + (size_t)b * rowsPerB * 128 + f;
  float a = 0.f, q = 0.f;
  for (int r = 0; r < rowsPerB; ++r) { float v = p[(size_t)r * 128]; a += v; q += v * v; }
  s[i] = a; s2[i] = q;
}

__global__ __launch_bounds__(128) void k_pairbn(const float* __restrict__ su, const float* __restrict__ su2,
                                                const float* __restrict__ sv, const float* __restrict__ sv2,
                                                const float* __restrict__ g, const float* __restrict__ bt,
                                                float* __restrict__ scale, float* __restrict__ shift) {
  int f = threadIdx.x;
  if (f >= 128) return;
  float S1 = 0.f, S2 = 0.f;
  for (int b = 0; b < B_; ++b) {
    float u = su[b * 128 + f], uu = su2[b * 128 + f];
    float v = sv[b * 128 + f], vv = sv2[b * 128 + f];
    S1 += (float)NT_ * u + (float)NL_ * v;
    S2 += (float)NT_ * uu + 2.f * u * v + (float)NL_ * vv;   // sum over pairs of (U+V)^2
  }
  const float P = (float)B_ * NL_ * NT_;
  float m = S1 / P, var = S2 / P - m * m, rs = rsqrtf(var + 1e-5f);
  scale[f] = g[f] * rs;
  shift[f] = bt[f] - m * scale[f];   // mlp_b cancels under BN
}

// ---------------- min distance ----------------
__global__ __launch_bounds__(256) void k_dist(const float* __restrict__ lp,
                                              const float* __restrict__ pp,
                                              float* __restrict__ dist) {
  int i = blockIdx.x * 256 + threadIdx.x;
  if (i >= B_ * NL_ * NT_) return;
  int b = i / (NL_ * NT_), r = i % (NL_ * NT_);
  int li = r / NT_, tj = r % NT_;
  const float* x = lp + (size_t)(b * NL_ + li) * 3;
  float x0 = x[0], x1 = x[1], x2 = x[2];
  float xs = x0 * x0 + x1 * x1 + x2 * x2;
  const float* y = pp + (size_t)(b * NT_ + tj) * APR_ * 3;
  float best = 1e30f;
  for (int a = 0; a < APR_; ++a) {
    float y0 = y[3 * a], y1 = y[3 * a + 1], y2 = y[3 * a + 2];
    float d2 = -2.f * (x0 * y0 + x1 * y1 + x2 * y2) + (y0 * y0 + y1 * y1 + y2 * y2) + xs;
    float d = (d2 < 0.f) ? 10000.f : sqrtf(d2);   // sqrt(neg)->NaN->10000 path
    best = fminf(best, d);
  }
  dist[i] = best;
}

// ---------------- fused pair MLP + MDN heads + loss accumulation ----------------
// block = one (b, 16-i, 16-j) tile = 256 pairs; cm built on the fly as WMMA A-fragments.
__global__ __launch_bounds__(256) void k_pair(const float* __restrict__ U, const float* __restrict__ V,
                                              const _Float16* __restrict__ HW,
                                              const float* __restrict__ scale, const float* __restrict__ shift,
                                              const float* __restrict__ pib, const float* __restrict__ sgb,
                                              const float* __restrict__ mub, const float* __restrict__ dist,
                                              float* __restrict__ acc) {
  __shared__ float sU[16 * 128], sV[16 * 128], sS[128], sH[128];
  __shared__ float sD[8][16][33];
  int tid = threadIdx.x, lane = tid & 31, wv = tid >> 5;
  int hf = lane >> 4, l16 = lane & 15;
  int blk = blockIdx.x;
  int b = blk / 75, rr = blk % 75, it = rr / 25, jt = rr % 25;
  for (int t = tid; t < 2048; t += 256) {
    int rw = t >> 7, f = t & 127;
    sU[t] = U[(size_t)(b * NL_ + it * 16 + rw) * 128 + f];
    sV[t] = V[(size_t)(b * NT_ + jt * 16 + rw) * 128 + f];
  }
  if (tid < 128) { sS[tid] = scale[tid]; sH[tid] = shift[tid]; }
  __syncthreads();

  float lr_nll = 0.f, lr_cnt = 0.f, lr_prob = 0.f;
  for (int ii = 0; ii < 2; ++ii) {
    int i = wv * 2 + ii;   // local ligand index (M-tile); pair row within WMMA tile = j
    v8f a0 = {}, a1 = {};
#pragma unroll
    for (int k0 = 0; k0 < 128; k0 += 32) {
      v16h A;
#pragma unroll
      for (int e = 0; e < 16; ++e) {
        int k = k0 + 16 * hf + e;
        float t = sU[i * 128 + k] + sV[l16 * 128 + k];
        t = t * sS[k] + sH[k];
        t = (t > 0.f) ? t : (expf(t) - 1.f);   // elu(bn(...))
        A[e] = (_Float16)t;
      }
      v16h b0 = *(const v16h*)(HW + (size_t)l16 * 128 + k0 + 16 * hf);
      v16h b1 = *(const v16h*)(HW + (size_t)(16 + l16) * 128 + k0 + 16 * hf);
      a0 = __builtin_amdgcn_wmma_f32_16x16x32_f16(false, A, false, b0, (short)0, a0, false, false);
      a1 = __builtin_amdgcn_wmma_f32_16x16x32_f16(false, A, false, b1, (short)0, a1, false, false);
    }
#pragma unroll
    for (int r = 0; r < 8; ++r) {
      sD[wv][8 * hf + r][l16] = a0[r];
      sD[wv][8 * hf + r][16 + l16] = a1[r];
    }
    __syncthreads();
    if (hf == 0) {
      int j = l16;
      float zp[10];
      float mx = -1e30f;
#pragma unroll
      for (int t = 0; t < 10; ++t) { zp[t] = sD[wv][j][t] + pib[t]; mx = fmaxf(mx, zp[t]); }
      float se = 0.f;
#pragma unroll
      for (int t = 0; t < 10; ++t) { zp[t] = expf(zp[t] - mx); se += zp[t]; }
      float y = dist[(size_t)(b * NL_ + it * 16 + i) * NT_ + jt * 16 + j];
      float sp = 0.f;
#pragma unroll
      for (int t = 0; t < 10; ++t) {
        float pi = zp[t] / se;
        float sg = sD[wv][j][10 + t] + sgb[t]; sg = ((sg > 0.f) ? sg : (expf(sg) - 1.f)) + 1.1f;
        float mu = sD[wv][j][20 + t] + mub[t]; mu = ((mu > 0.f) ? mu : (expf(mu) - 1.f)) + 1.0f;
        float dd = (y - mu) / sg;
        float ll = -0.5f * dd * dd - logf(sg) - 0.5f * LOG2PI_;
        sp += (pi + 1e-10f) * expf(ll);
      }
      float nll = -logf(sp + 1e-10f);
      if (y <= 7.0f) { lr_nll += nll; lr_cnt += 1.f; }
      lr_prob += sp;
    }
    __syncthreads();
  }
  // block reduce (reuse sU)
  sU[tid] = lr_nll; sU[256 + tid] = lr_cnt; sU[512 + tid] = lr_prob;
  __syncthreads();
  for (int o = 128; o > 0; o >>= 1) {
    if (tid < o) { sU[tid] += sU[tid + o]; sU[256 + tid] += sU[256 + tid + o]; sU[512 + tid] += sU[512 + tid + o]; }
    __syncthreads();
  }
  if (tid == 0) {
    atomicAdd(&acc[0], sU[0]);
    atomicAdd(&acc[1], sU[256]);
    atomicAdd(&acc[4 + b], sU[512]);
  }
}

// ---------------- auxiliary CE heads ----------------
__global__ __launch_bounds__(128) void k_atom(const float* __restrict__ hl, const float* __restrict__ lx,
                                              const float* __restrict__ w, const float* __restrict__ bb,
                                              float* __restrict__ acc) {
  int n = blockIdx.x * 128 + threadIdx.x;
  if (n >= NLT_) return;
  int lab = 0; float bv = lx[n * 41];
  for (int c = 1; c < 17; ++c) { float v = lx[n * 41 + c]; if (v > bv) { bv = v; lab = c; } }
  float lg[17];
  for (int c = 0; c < 17; ++c) lg[c] = bb[c];
  for (int k = 0; k < 128; ++k) {
    float h = hl[n * 128 + k];
    for (int c = 0; c < 17; ++c) lg[c] += h * w[k * 17 + c];
  }
  float mx = lg[0];
  for (int c = 1; c < 17; ++c) mx = fmaxf(mx, lg[c]);
  float se = 0.f;
  for (int c = 0; c < 17; ++c) se += expf(lg[c] - mx);
  atomicAdd(&acc[2], -(lg[lab] - mx - logf(se)));
}

__global__ __launch_bounds__(128) void k_bond(const float* __restrict__ hl, const float* __restrict__ lea,
                                              const int* __restrict__ ei, const float* __restrict__ w,
                                              const float* __restrict__ bb, float* __restrict__ acc) {
  int e = blockIdx.x * 128 + threadIdx.x;
  if (e >= LE_) return;
  int lab = 0; float bv = lea[e * 10];
  for (int c = 1; c < 4; ++c) { float v = lea[e * 10 + c]; if (v > bv) { bv = v; lab = c; } }
  int s = ei[e], d = ei[LE_ + e];
  float lg[4] = { bb[0], bb[1], bb[2], bb[3] };
  for (int k = 0; k < 128; ++k) {
    float h = hl[s * 128 + k];
    for (int c = 0; c < 4; ++c) lg[c] += h * w[k * 4 + c];
  }
  for (int k = 0; k < 128; ++k) {
    float h = hl[d * 128 + k];
    for (int c = 0; c < 4; ++c) lg[c] += h * w[(128 + k) * 4 + c];
  }
  float mx = fmaxf(fmaxf(lg[0], lg[1]), fmaxf(lg[2], lg[3]));
  float se = 0.f;
  for (int c = 0; c < 4; ++c) se += expf(lg[c] - mx);
  atomicAdd(&acc[3], -(lg[lab] - mx - logf(se)));
}

__global__ void k_final(const float* __restrict__ acc, const float* __restrict__ labels,
                        float* __restrict__ out) {
  if (threadIdx.x != 0 || blockIdx.x != 0) return;
  float mdn = acc[0] / fmaxf(acc[1], 1.f);
  float ace = acc[2] / (float)NLT_;
  float bce = acc[3] / (float)LE_;
  float ym = 0.f, lm = 0.f;
  for (int b = 0; b < 16; ++b) { ym += acc[4 + b]; lm += labels[b]; }
  ym /= 16.f; lm /= 16.f;
  float nu = 0.f, d1 = 0.f, d2 = 0.f;
  for (int b = 0; b < 16; ++b) {
    float a = acc[4 + b] - ym, c = labels[b] - lm;
    nu += a * c; d1 += a * a; d2 += c * c;
  }
  float affi = nu / (sqrtf(d1) * sqrtf(d2) + 1e-12f);
  float loss = mdn + 0.f * affi + 1e-3f * (ace + bce);
  out[0] = loss; out[1] = mdn; out[2] = affi; out[3] = ace; out[4] = bce;
  for (int b = 0; b < 16; ++b) out[5 + b] = acc[4 + b];
}

// ---------------- host orchestration ----------------
extern "C" void kernel_launch(void* const* d_in, const int* in_sizes, int n_in,
                              void* d_out, int out_size, void* d_ws, size_t ws_size,
                              hipStream_t stream) {
  (void)in_sizes; (void)n_in; (void)out_size; (void)ws_size;
  const float* lig_x     = (const float*)d_in[0];
  const float* lig_ea    = (const float*)d_in[1];
  const float* lig_pos   = (const float*)d_in[2];
  const float* prot_x    = (const float*)d_in[3];
  const float* prot_pos  = (const float*)d_in[5];
  const float* labels    = (const float*)d_in[6];
  const float* lig_enc_w = (const float*)d_in[7];   const float* lig_enc_b  = (const float*)d_in[8];
  const float* lig_eenc_w= (const float*)d_in[9];   const float* lig_eenc_b = (const float*)d_in[10];
  const float* lig_W     = (const float*)d_in[11];  const float* lig_bw     = (const float*)d_in[12];
  const float* lig_gamma = (const float*)d_in[13];  const float* lig_beta   = (const float*)d_in[14];
  const float* prot_enc_w= (const float*)d_in[15];  const float* prot_enc_b = (const float*)d_in[16];
  const float* prot_eenc_w=(const float*)d_in[17];  const float* prot_eenc_b= (const float*)d_in[18];
  const float* prot_W    = (const float*)d_in[19];  const float* prot_bw    = (const float*)d_in[20];
  const float* prot_gamma= (const float*)d_in[21];  const float* prot_beta  = (const float*)d_in[22];
  const float* prot_ea   = (const float*)d_in[4];
  const float* mlp_w     = (const float*)d_in[23];
  const float* mlp_gamma = (const float*)d_in[25];  const float* mlp_beta   = (const float*)d_in[26];
  const float* pi_w      = (const float*)d_in[27];  const float* pi_b       = (const float*)d_in[28];
  const float* sigma_w   = (const float*)d_in[29];  const float* sigma_b    = (const float*)d_in[30];
  const float* mu_w      = (const float*)d_in[31];  const float* mu_b       = (const float*)d_in[32];
  const float* atom_w    = (const float*)d_in[33];  const float* atom_b     = (const float*)d_in[34];
  const float* bond_w    = (const float*)d_in[35];  const float* bond_b     = (const float*)d_in[36];
  const int*   lig_ei    = (const int*)d_in[37];
  const int*   prot_ei   = (const int*)d_in[38];
  float* out = (float*)d_out;

  size_t cur = 0;
  auto alloc = [&](size_t bytes) -> char* {
    char* p = (char*)d_ws + cur;
    cur += (bytes + 255) & ~(size_t)255;
    return p;
  };
  float* hl   = (float*)alloc((size_t)NLT_ * 128 * 4);
  float* el   = (float*)alloc((size_t)LE_  * 128 * 4);
  float* ht   = (float*)alloc((size_t)NTT_ * 128 * 4);
  float* et   = (float*)alloc((size_t)PE_  * 128 * 4);
  float* tA   = (float*)alloc((size_t)NTT_ * 128 * 4);
  float* tB   = (float*)alloc((size_t)NTT_ * 128 * 4);
  float* tD   = (float*)alloc((size_t)NTT_ * 128 * 4);
  float* tE   = (float*)alloc((size_t)NTT_ * 128 * 4);
  float* tC   = (float*)alloc((size_t)PE_  * 128 * 4);
  float* numb = (float*)alloc((size_t)NTT_ * 128 * 4);
  float* denb = (float*)alloc((size_t)NTT_ * 128 * 4);
  float* Ub   = (float*)alloc((size_t)NLT_ * 128 * 4);
  float* Vb   = (float*)alloc((size_t)NTT_ * 128 * 4);
  float* dst  = (float*)alloc((size_t)B_ * NL_ * NT_ * 4);
  float* bnM  = (float*)alloc(128 * 4);
  float* bnR  = (float*)alloc(128 * 4);
  float* pSc  = (float*)alloc(128 * 4);
  float* pSh  = (float*)alloc(128 * 4);
  float* su   = (float*)alloc(2048 * 4);
  float* su2  = (float*)alloc(2048 * 4);
  float* sv   = (float*)alloc(2048 * 4);
  float* sv2  = (float*)alloc(2048 * 4);
  float* zer  = (float*)alloc(128 * 4);
  float* acc  = (float*)alloc(32 * 4);
  _Float16* WtL = (_Float16*)alloc((size_t)30 * 16384 * 2);
  _Float16* WtP = (_Float16*)alloc((size_t)30 * 16384 * 2);
  _Float16* WtU = (_Float16*)alloc((size_t)16384 * 2);
  _Float16* WtV = (_Float16*)alloc((size_t)16384 * 2);
  _Float16* HW  = (_Float16*)alloc((size_t)4096 * 2);

  // init accumulators + zero bias
  k_fill0<<<1, 256, 0, stream>>>(acc, 32);
  k_fill0<<<1, 256, 0, stream>>>(zer, 128);

  // weight prep (transpose + f16)
  k_prep_wt<<<(30 * 16384 + 255) / 256, 256, 0, stream>>>(lig_W, WtL, 30 * 16384);
  k_prep_wt<<<(30 * 16384 + 255) / 256, 256, 0, stream>>>(prot_W, WtP, 30 * 16384);
  k_prep_wt<<<(16384 + 255) / 256, 256, 0, stream>>>(mlp_w, WtU, 16384);
  k_prep_wt<<<(16384 + 255) / 256, 256, 0, stream>>>(mlp_w + 16384, WtV, 16384);
  k_prep_head<<<(32 * 128 + 255) / 256, 256, 0, stream>>>(pi_w, sigma_w, mu_w, HW);

  // encoders
  k_enc<<<(NLT_ * 128 + 255) / 256, 256, 0, stream>>>(lig_x, lig_enc_w, lig_enc_b, hl, NLT_, 41);
  k_enc<<<(LE_  * 128 + 255) / 256, 256, 0, stream>>>(lig_ea, lig_eenc_w, lig_eenc_b, el, LE_, 10);
  k_enc<<<(NTT_ * 128 + 255) / 256, 256, 0, stream>>>(prot_x, prot_enc_w, prot_enc_b, ht, NTT_, 41);
  k_enc<<<(PE_  * 128 + 255) / 256, 256, 0, stream>>>(prot_ea, prot_eenc_w, prot_eenc_b, et, PE_, 5);

  auto run_ggcn = [&](float* x, float* e, int N, int E, const _Float16* Wt, const float* bw,
                      const float* gamma, const float* beta, const int* ei) {
    for (int l = 0; l < 6; ++l) {
      k_gemm128<<<N / 16, 256, 0, stream>>>(x, Wt + (size_t)(l * 5 + 0) * 16384, bw + (l * 5 + 0) * 128, tA);
      k_gemm128<<<N / 16, 256, 0, stream>>>(x, Wt + (size_t)(l * 5 + 1) * 16384, bw + (l * 5 + 1) * 128, tB);
      k_gemm128<<<E / 16, 256, 0, stream>>>(e, Wt + (size_t)(l * 5 + 2) * 16384, bw + (l * 5 + 2) * 128, tC);
      k_gemm128<<<N / 16, 256, 0, stream>>>(x, Wt + (size_t)(l * 5 + 3) * 16384, bw + (l * 5 + 3) * 128, tD);
      k_gemm128<<<N / 16, 256, 0, stream>>>(x, Wt + (size_t)(l * 5 + 4) * 16384, bw + (l * 5 + 4) * 128, tE);
      k_fill0<<<(N * 128 + 255) / 256, 256, 0, stream>>>(numb, N * 128);
      k_fill0<<<(N * 128 + 255) / 256, 256, 0, stream>>>(denb, N * 128);
      k_edge<<<(E * 128 + 255) / 256, 256, 0, stream>>>(tB, tD, tE, tC, ei, E, numb, denb);
      k_combine<<<(N * 128 + 255) / 256, 256, 0, stream>>>(tA, numb, denb, N * 128);
      k_bnstats<<<128, 256, 0, stream>>>(tA, N, bnM, bnR);
      k_bnres<<<(N * 128 + 255) / 256, 256, 0, stream>>>(x, tA, bnM, bnR,
          gamma + (l * 2 + 0) * 128, beta + (l * 2 + 0) * 128, N * 128);
      k_bnstats<<<128, 256, 0, stream>>>(tC, E, bnM, bnR);
      k_bnres<<<(E * 128 + 255) / 256, 256, 0, stream>>>(e, tC, bnM, bnR,
          gamma + (l * 2 + 1) * 128, beta + (l * 2 + 1) * 128, E * 128);
    }
  };
  run_ggcn(hl, el, NLT_, LE_, WtL, lig_bw, lig_gamma, lig_beta, lig_ei);
  run_ggcn(ht, et, NTT_, PE_, WtP, prot_bw, prot_gamma, prot_beta, prot_ei);

  // pair-MLP decomposition: U = hl @ mlp_w[:128], V = ht @ mlp_w[128:]
  k_gemm128<<<NLT_ / 16, 256, 0, stream>>>(hl, WtU, zer, Ub);
  k_gemm128<<<NTT_ / 16, 256, 0, stream>>>(ht, WtV, zer, Vb);

  // analytic pair BN stats
  k_colsum<<<(B_ * 128 + 255) / 256, 256, 0, stream>>>(Ub, su, su2, NL_);
  k_colsum<<<(B_ * 128 + 255) / 256, 256, 0, stream>>>(Vb, sv, sv2, NT_);
  k_pairbn<<<1, 128, 0, stream>>>(su, su2, sv, sv2, mlp_gamma, mlp_beta, pSc, pSh);

  // distances + fused MDN
  k_dist<<<(B_ * NL_ * NT_ + 255) / 256, 256, 0, stream>>>(lig_pos, prot_pos, dst);
  k_pair<<<B_ * (NL_ / 16) * (NT_ / 16), 256, 0, stream>>>(Ub, Vb, HW, pSc, pSh,
                                                           pi_b, sigma_b, mu_b, dst, acc);

  // aux heads + finalize
  k_atom<<<(NLT_ + 127) / 128, 128, 0, stream>>>(hl, lig_x, atom_w, atom_b, acc);
  k_bond<<<(LE_ + 127) / 128, 128, 0, stream>>>(hl, lig_ea, lig_ei, bond_w, bond_b, acc);
  k_final<<<1, 32, 0, stream>>>(acc, labels, out);
}
